// MAPBlock_17514876634168
// MI455X (gfx1250) — matrix operand order repaired and checked by reading, not compiled, not run on hardware
//
#include <hip/hip_runtime.h>
#include <hip/hip_bf16.h>

// ---------------------------------------------------------------------------
// MAP block (single-latent attention pooling), B=32 N=4096 C=768 H=24 D=32.
// Key fusions: kv = x@(proj_w@kv_w); scores = x @ (Wk@q) (K never formed);
// flash softmax per 128-row chunk fused with the bf16 WMMA V-GEMM so kv is
// never written to HBM. HBM floor ~= one read of x (402MB) ~= 17us @23.3TB/s.
// ---------------------------------------------------------------------------

typedef __attribute__((ext_vector_type(16))) __bf16 bf16x16;
typedef __attribute__((ext_vector_type(8)))  float  f32x8;

#define CC   768
#define NN   4096
#define BB   32
#define HH   24
#define DD   32
#define HID  3072
#define NT   128          // rows of x per workgroup chunk
#define NCH  (NN / NT)    // 32 chunks per batch
#define XST  776          // LDS row stride in bf16 halves (768 + 8 pad -> no bank conflicts)
#define SST  36           // LDS score row stride in floats

__device__ __forceinline__ unsigned short f2bf(float f) {
  unsigned int u = __float_as_uint(f);
  u += 0x7fffu + ((u >> 16) & 1u);       // round-to-nearest-even
  return (unsigned short)(u >> 16);
}

// Load a 32-byte WMMA fragment as two 16B chunks separated by `off` halves.
// A-frag (16x32 bf16): per half-wave K {0..7,16..23}/{8..15,24..31} -> off=16.
// B-frag (32x16 bf16): same K split as A, {0..7,16..23}/{8..15,24..31} -> off=16.
__device__ __forceinline__ bf16x16 ld_frag(const unsigned short* p, int off) {
  union { bf16x16 v; uint4 q[2]; } u;
  u.q[0] = *(const uint4*)(p);
  u.q[1] = *(const uint4*)(p + off);
  return u.v;
}

__device__ __forceinline__ f32x8 wmma_bf16(bf16x16 a, bf16x16 b, f32x8 c) {
  return __builtin_amdgcn_wmma_f32_16x16x32_bf16(false, a, false, b, (short)0, c,
                                                 false, false);
}

// ------------------------------ prep kernels -------------------------------

__global__ void k_lat(const float* __restrict__ latents,
                      const float* __restrict__ init_embed,
                      float* __restrict__ lat) {
  int i = blockIdx.x * 256 + threadIdx.x;
  if (i < BB * CC) lat[i] = latents[i % CC] + init_embed[i];
}

// qs[b,c] = scale * (lat[b,:] @ q_w[:,c])
__global__ __launch_bounds__(768) void k_qs(const float* __restrict__ lat,
                                            const float* __restrict__ q_w,
                                            float* __restrict__ qs) {
  int b = blockIdx.x, c = threadIdx.x;
  __shared__ float lsh[CC];
  lsh[c] = lat[b * CC + c];
  __syncthreads();
  float a = 0.f;
  for (int k = 0; k < CC; k++) a = fmaf(lsh[k], q_w[(long)k * CC + c], a);
  qs[b * CC + c] = a * 0.17677669529663689f;  // D^-0.5
}

__global__ void k_pwb(const float* __restrict__ proj_w, unsigned short* __restrict__ pwb) {
  int i = blockIdx.x * 256 + threadIdx.x;  // grid covers 768*768
  pwb[i] = f2bf(proj_w[i]);
}

// T[bh, m] = sum_d kv_w[m, h*D+d] * qs[b, h*D+d]   (bf16 out, A-matrix of GEMM1)
__global__ void k_t(const float* __restrict__ kv_w, const float* __restrict__ qs,
                    unsigned short* __restrict__ T) {
  int bh = blockIdx.x, b = bh / HH, h = bh % HH;
  __shared__ float qsh[DD];
  if (threadIdx.x < DD) qsh[threadIdx.x] = qs[b * CC + h * DD + threadIdx.x];
  __syncthreads();
  for (int m = threadIdx.x; m < CC; m += 256) {
    const float* row = kv_w + (long)m * (2 * CC) + h * DD;
    float a = 0.f;
#pragma unroll
    for (int d = 0; d < DD; d++) a = fmaf(row[d], qsh[d], a);
    T[(long)bh * CC + m] = f2bf(a);
  }
}

// kvvT[j, m] = kv_w[m, C+j]  (bf16 transpose of the V half, A-matrix of GEMM2)
__global__ void k_kvvT(const float* __restrict__ kv_w, unsigned short* __restrict__ kvvT) {
  __shared__ float t[32][33];
  int jt = blockIdx.x * 32, mt = blockIdx.y * 32;
  int tx = threadIdx.x, ty = threadIdx.y;
#pragma unroll
  for (int i = 0; i < 4; i++) {
    int m = mt + ty + i * 8;
    t[ty + i * 8][tx] = kv_w[(long)m * (2 * CC) + CC + jt + tx];
  }
  __syncthreads();
#pragma unroll
  for (int i = 0; i < 4; i++) {
    int j = jt + ty + i * 8;
    kvvT[(long)j * CC + mt + tx] = f2bf(t[tx][ty + i * 8]);
  }
}

// b2v[j] = proj_b @ kv_w[:, C+j]
__global__ void k_b2v(const float* __restrict__ proj_b, const float* __restrict__ kv_w,
                      float* __restrict__ b2v) {
  int j = blockIdx.x * 256 + threadIdx.x;
  float a = 0.f;
  for (int m = 0; m < CC; m++) a = fmaf(proj_b[m], kv_w[(long)m * (2 * CC) + CC + j], a);
  b2v[j] = a;
}

// D(M,N) = A(M,K) @ Bsrc(N,K)^T  with M=N=K=768, bf16 in/out, f32 accum.
// pad24: remap output row m -> (m/24)*32 + m%24 (Q2p padding, rows 24..31 stay 0).
__global__ __launch_bounds__(256) void k_gemm_nt(const unsigned short* __restrict__ A,
                                                 const unsigned short* __restrict__ Bsrc,
                                                 unsigned short* __restrict__ Dout,
                                                 int pad24) {
  const int tid = threadIdx.x, lane = tid & 31, w = tid >> 5;
  const int g = lane >> 4, lr = lane & 15;
  const int m0 = blockIdx.y * 64 + (w >> 1) * 16;
  const int n0 = blockIdx.x * 32 + (w & 1) * 16;
  f32x8 acc = {};
  const unsigned short* arow = A + (long)(m0 + lr) * CC + g * 8;
  const unsigned short* brow = Bsrc + (long)(n0 + lr) * CC + g * 8;
  for (int k0 = 0; k0 < CC; k0 += 32) {
    bf16x16 a = ld_frag(arow + k0, 16);
    bf16x16 bq = ld_frag(brow + k0, 16);
    acc = wmma_bf16(a, bq, acc);
  }
  const int col = n0 + lr;
#pragma unroll
  for (int v = 0; v < 8; v++) {
    int m = m0 + v + 8 * g;
    int orow = pad24 ? ((m / 24) * 32 + (m % 24)) : m;
    Dout[(long)orow * CC + col] = f2bf(acc[v]);
  }
}

// ------------------------- fused V-GEMM + flash attn -----------------------
// grid(NCH, BB), 512 threads (16 waves). Dynamic LDS:
//   Xs: 128 x 776 bf16 (x tile)   Ss: 128 x 36 f32 (scores -> p)   msh: 32 f32
__global__ __launch_bounds__(512) void k_main(const float* __restrict__ x,
                                              const unsigned short* __restrict__ Q2p,
                                              const unsigned short* __restrict__ WvT,
                                              float* __restrict__ part_m,
                                              float* __restrict__ part_s,
                                              float* __restrict__ part_acc) {
  extern __shared__ char smem[];
  unsigned short* Xs = (unsigned short*)smem;
  float* Ss = (float*)(smem + NT * XST * 2);
  float* msh = Ss + NT * SST;

  const int tid = threadIdx.x;
  const int chunk = blockIdx.x, b = blockIdx.y;
  const int n0 = chunk * NT;
  const int lane = tid & 31, w = tid >> 5;
  const int g = lane >> 4, lr = lane & 15;

  // ---- Phase A: load x tile, convert fp32 -> bf16 into LDS ----
  const float4* xrow = (const float4*)(x + ((long)b * NN + n0) * CC);
  for (int i = tid; i < NT * (CC / 4); i += 512) {
    int r = i / (CC / 4), c4 = i % (CC / 4);
    float4 v = xrow[(long)r * (CC / 4) + c4];
    ushort4 h;
    h.x = f2bf(v.x); h.y = f2bf(v.y); h.z = f2bf(v.z); h.w = f2bf(v.w);
    *(ushort4*)(Xs + r * XST + c4 * 4) = h;
  }
  __syncthreads();

  // ---- Phase B: scores tile (128 x 32) = Xtile @ Q2p[b] ----
  {
    const int rt = w >> 1, ctile = w & 1;
    f32x8 acc = {};
    const unsigned short* arow = Xs + (rt * 16 + lr) * XST + g * 8;
    const unsigned short* brow = Q2p + (long)(b * 32 + ctile * 16 + lr) * CC + g * 8;
    for (int k0 = 0; k0 < CC; k0 += 32) {
      bf16x16 a = ld_frag(arow + k0, 16);
      bf16x16 bq = ld_frag(brow + k0, 16);
      acc = wmma_bf16(a, bq, acc);
    }
#pragma unroll
    for (int v = 0; v < 8; v++)
      Ss[(rt * 16 + v + 8 * g) * SST + ctile * 16 + lr] = acc[v];
  }
  __syncthreads();

  // ---- Phase C: chunk-local softmax stats ----
  if (w == 0 && lane < HH) {
    float m = -3.0e38f;
    for (int n = 0; n < NT; n++) m = fmaxf(m, Ss[n * SST + lane]);
    msh[lane] = m;
  }
  __syncthreads();
  for (int i = tid; i < NT * 32; i += 512) {
    int n = i >> 5, h = i & 31;
    Ss[n * SST + h] = (h < HH) ? __expf(Ss[n * SST + h] - msh[h]) : 0.0f;
  }
  __syncthreads();
  if (w == 0 && lane < HH) {
    float s = 0.f;
    for (int n = 0; n < NT; n++) s += Ss[n * SST + lane];
    int pi = (b * NCH + chunk) * HH + lane;
    part_m[pi] = msh[lane];
    part_s[pi] = s;
  }

  // ---- Phase D: V = Xtile @ Wv (wave owns 48 cols), then P.V on fragments ----
  {
    const int jbase0 = w * 48;
    for (int ct = 0; ct < 3; ct++) {
      const int jb = jbase0 + ct * 16;
      const int h = jb >> 5;  // head is constant across a 16-col tile
      f32x8 acc[8];
#pragma unroll
      for (int r = 0; r < 8; r++) acc[r] = (f32x8){};
      const unsigned short* brow = WvT + (long)(jb + lr) * CC + g * 8;
      for (int k0 = 0; k0 < CC; k0 += 32) {
        bf16x16 bw = ld_frag(brow + k0, 16);
#pragma unroll
        for (int r = 0; r < 8; r++) {
          bf16x16 a = ld_frag(Xs + (r * 16 + lr) * XST + g * 8 + k0, 16);
          acc[r] = wmma_bf16(a, bw, acc[r]);
        }
      }
      // out[j] += sum_n p[n,h] * V[n,j]; lane holds col j = jb+lr, rows v+8g of each tile
      float vs = 0.f;
#pragma unroll
      for (int r = 0; r < 8; r++) {
#pragma unroll
        for (int v = 0; v < 8; v++)
          vs = fmaf(Ss[(r * 16 + v + 8 * g) * SST + h], acc[r][v], vs);
      }
      vs += __shfl_down(vs, 16, 32);  // combine the two half-wave row groups
      if (lane < 16)
        part_acc[(long)(b * NCH + chunk) * CC + jb + lr] = vs;
    }
  }
}

// ------------------------------ epilogue -----------------------------------

// Combine per-chunk flash partials -> vals[b, :] (+ V bias once; softmax sums to 1)
__global__ __launch_bounds__(768) void k_reduce(const float* __restrict__ part_m,
                                                const float* __restrict__ part_s,
                                                const float* __restrict__ part_acc,
                                                const float* __restrict__ b2v,
                                                float* __restrict__ vals) {
  int b = blockIdx.x, tid = threadIdx.x;
  __shared__ float Msh[HH], Ssh[HH];
  if (tid < HH) {
    float M = -3.0e38f;
    for (int ch = 0; ch < NCH; ch++) M = fmaxf(M, part_m[(b * NCH + ch) * HH + tid]);
    float S = 0.f;
    for (int ch = 0; ch < NCH; ch++)
      S += part_s[(b * NCH + ch) * HH + tid] * __expf(part_m[(b * NCH + ch) * HH + tid] - M);
    Msh[tid] = M; Ssh[tid] = S;
  }
  __syncthreads();
  int h = tid >> 5;
  float a = 0.f;
  for (int ch = 0; ch < NCH; ch++)
    a = fmaf(__expf(part_m[(b * NCH + ch) * HH + h] - Msh[h]),
             part_acc[(long)(b * NCH + ch) * CC + tid], a);
  vals[b * CC + tid] = a / Ssh[h] + b2v[tid];
}

// lat2 = rmsnorm(lat + vals@attn_o_w + attn_o_b, attn_norm_w)
__global__ __launch_bounds__(768) void k_attn(const float* __restrict__ vals,
                                              const float* __restrict__ attn_o_w,
                                              const float* __restrict__ attn_o_b,
                                              const float* __restrict__ lat,
                                              const float* __restrict__ attn_norm_w,
                                              float* __restrict__ lat2) {
  int b = blockIdx.x, tid = threadIdx.x;
  __shared__ float vsh[CC];
  __shared__ float red[CC];
  vsh[tid] = vals[b * CC + tid];
  __syncthreads();
  float y = attn_o_b[tid] + lat[b * CC + tid];
  for (int k = 0; k < CC; k++) y = fmaf(vsh[k], attn_o_w[(long)k * CC + tid], y);
  red[tid] = y * y;
  __syncthreads();
  if (tid < 256) red[tid] += red[tid + 256] + red[tid + 512];
  __syncthreads();
  for (int s = 128; s > 0; s >>= 1) {
    if (tid < s) red[tid] += red[tid + s];
    __syncthreads();
  }
  float sc = rsqrtf(red[0] * (1.0f / CC) + 1e-6f);
  lat2[b * CC + tid] = y * sc * attn_norm_w[tid];
}

// hidden = projected * silu(gate), where [projected|gate] = lat2@glu_w + glu_b
__global__ __launch_bounds__(256) void k_glu(const float* __restrict__ lat2,
                                             const float* __restrict__ glu_w,
                                             const float* __restrict__ glu_b,
                                             float* __restrict__ hidden) {
  int b = blockIdx.y, jj = blockIdx.x * 256 + threadIdx.x;
  __shared__ float lsh[CC];
  for (int i = threadIdx.x; i < CC; i += 256) lsh[i] = lat2[b * CC + i];
  __syncthreads();
  float p = glu_b[jj], gt = glu_b[HID + jj];
  for (int k = 0; k < CC; k++) {
    float l = lsh[k];
    p  = fmaf(l, glu_w[(long)k * (2 * HID) + jj], p);
    gt = fmaf(l, glu_w[(long)k * (2 * HID) + HID + jj], gt);
  }
  hidden[(long)b * HID + jj] = p * (gt / (1.0f + __expf(-gt)));
}

// out = rmsnorm(lat2 + hidden@mlp_out_w + mlp_out_b, mlp_norm_w)
__global__ __launch_bounds__(768) void k_mlp(const float* __restrict__ hidden,
                                             const float* __restrict__ mlp_out_w,
                                             const float* __restrict__ mlp_out_b,
                                             const float* __restrict__ lat2,
                                             const float* __restrict__ mlp_norm_w,
                                             float* __restrict__ out) {
  int b = blockIdx.x, tid = threadIdx.x;
  __shared__ float hsh[HID];
  __shared__ float red[CC];
  for (int i = tid; i < HID; i += 768) hsh[i] = hidden[(long)b * HID + i];
  __syncthreads();
  float z = mlp_out_b[tid];
  for (int k = 0; k < HID; k++) z = fmaf(hsh[k], mlp_out_w[(long)k * CC + tid], z);
  z += lat2[b * CC + tid];
  red[tid] = z * z;
  __syncthreads();
  if (tid < 256) red[tid] += red[tid + 256] + red[tid + 512];
  __syncthreads();
  for (int s = 128; s > 0; s >>= 1) {
    if (tid < s) red[tid] += red[tid + s];
    __syncthreads();
  }
  float sc = rsqrtf(red[0] * (1.0f / CC) + 1e-6f);
  out[b * CC + tid] = z * sc * mlp_norm_w[tid];
}

// ------------------------------ launcher -----------------------------------

extern "C" void kernel_launch(void* const* d_in, const int* in_sizes, int n_in,
                              void* d_out, int out_size, void* d_ws, size_t ws_size,
                              hipStream_t stream) {
  const float* x          = (const float*)d_in[0];
  const float* init_embed = (const float*)d_in[1];
  const float* latents    = (const float*)d_in[2];
  const float* proj_w     = (const float*)d_in[3];
  const float* proj_b     = (const float*)d_in[4];
  const float* q_w        = (const float*)d_in[5];
  const float* kv_w       = (const float*)d_in[6];
  const float* attn_o_w   = (const float*)d_in[7];
  const float* attn_o_b   = (const float*)d_in[8];
  const float* attn_norm_w= (const float*)d_in[9];
  const float* mlp_norm_w = (const float*)d_in[10];
  const float* glu_w      = (const float*)d_in[11];
  const float* glu_b      = (const float*)d_in[12];
  const float* mlp_out_w  = (const float*)d_in[13];
  const float* mlp_out_b  = (const float*)d_in[14];
  float* out = (float*)d_out;

  // workspace carve-up (~10.4 MB total)
  char* ws = (char*)d_ws;
  constexpr size_t O_LAT  = 0;                         // B*C f32
  constexpr size_t O_QS   = O_LAT  + 98304;            // B*C f32
  constexpr size_t O_T    = O_QS   + 98304;            // 768*768 bf16
  constexpr size_t O_PWB  = O_T    + 1179648;          // 768*768 bf16
  constexpr size_t O_KVVT = O_PWB  + 1179648;          // 768*768 bf16
  constexpr size_t O_Q2P  = O_KVVT + 1179648;          // B*32*768 bf16 (padded)
  constexpr size_t O_WVT  = O_Q2P  + 1572864;          // 768*768 bf16
  constexpr size_t O_B2V  = O_WVT  + 1179648;          // 768 f32
  constexpr size_t O_PM   = O_B2V  + 3072;             // B*NCH*H f32
  constexpr size_t O_PS   = O_PM   + 98304;            // B*NCH*H f32
  constexpr size_t O_PA   = O_PS   + 98304;            // B*NCH*C f32
  constexpr size_t O_VALS = O_PA   + 3145728;          // B*C f32
  constexpr size_t O_LAT2 = O_VALS + 98304;            // B*C f32
  constexpr size_t O_HID  = O_LAT2 + 98304;            // B*HID f32

  float* lat   = (float*)(ws + O_LAT);
  float* qs    = (float*)(ws + O_QS);
  unsigned short* T    = (unsigned short*)(ws + O_T);
  unsigned short* pwb  = (unsigned short*)(ws + O_PWB);
  unsigned short* kvvT = (unsigned short*)(ws + O_KVVT);
  unsigned short* Q2p  = (unsigned short*)(ws + O_Q2P);
  unsigned short* WvT  = (unsigned short*)(ws + O_WVT);
  float* b2v   = (float*)(ws + O_B2V);
  float* pm    = (float*)(ws + O_PM);
  float* ps    = (float*)(ws + O_PS);
  float* pa    = (float*)(ws + O_PA);
  float* vals  = (float*)(ws + O_VALS);
  float* lat2  = (float*)(ws + O_LAT2);
  float* hid   = (float*)(ws + O_HID);

  // ---- prep (all tiny) ----
  k_lat<<<96, 256, 0, stream>>>(latents, init_embed, lat);
  k_qs<<<BB, 768, 0, stream>>>(lat, q_w, qs);
  k_pwb<<<2304, 256, 0, stream>>>(proj_w, pwb);
  k_t<<<BB * HH, 256, 0, stream>>>(kv_w, qs, T);
  k_kvvT<<<dim3(24, 24), dim3(32, 8), 0, stream>>>(kv_w, kvvT);
  hipMemsetAsync(Q2p, 0, 1572864, stream);
  // Q2p[b,h,:] = proj_w @ t[b,h,:]   (M=bh, N=kk, K=m) -> padded rows of 32
  k_gemm_nt<<<dim3(24, 12), 256, 0, stream>>>(T, pwb, Q2p, 1);
  // WvT[j,c] = sum_m kv_w[m,C+j] * proj_w[c,m]
  k_gemm_nt<<<dim3(24, 12), 256, 0, stream>>>(kvvT, pwb, WvT, 0);
  k_b2v<<<3, 256, 0, stream>>>(proj_b, kv_w, b2v);

  // ---- fused V-GEMM + flash attention ----
  constexpr int SMEM_MAIN = NT * XST * 2 + NT * SST * 4 + 32 * 4;  // ~217 KB of 320 KB WGP LDS
  hipFuncSetAttribute(reinterpret_cast<const void*>(&k_main),
                      hipFuncAttributeMaxDynamicSharedMemorySize, SMEM_MAIN);
  k_main<<<dim3(NCH, BB), 512, SMEM_MAIN, stream>>>(x, Q2p, WvT, pm, ps, pa);

  // ---- epilogue ----
  k_reduce<<<BB, 768, 0, stream>>>(pm, ps, pa, b2v, vals);
  k_attn<<<BB, 768, 0, stream>>>(vals, attn_o_w, attn_o_b, lat, attn_norm_w, lat2);
  k_glu<<<dim3(HID / 256, BB), 256, 0, stream>>>(lat2, glu_w, glu_b, hid);
  k_mlp<<<BB, 768, 0, stream>>>(hid, mlp_out_w, mlp_out_b, lat2, mlp_norm_w, out);
}
